// Conv2dWeightModulate_62972810494551
// MI455X (gfx1250) — matrix operand order repaired and checked by reading, not compiled, not run on hardware
//
#include <hip/hip_runtime.h>

// ---------------------------------------------------------------------------
// StyleGAN2 modulated conv (B=8, Cin=Cout=512, K=3, H=W=64), fp32 in/out.
// Compute-bound (155 GFLOP vs ~144MB traffic) -> bf16 WMMA f32-accumulate
// implicit GEMM: per batch  [512 x 4096] = [512 x 4608] x [4608 x 4096].
// GEMM-K is ordered k-major (kk = k*512 + i) so every 32-wide K-slab has a
// uniform (ky,kx) -> im2col address math is SALU; hot loop is wmma+ds+vmem.
// Block tile 128x128 (2 image rows), wave tile 32x64 = 2x4 WMMA tiles.
// Modulation folded into the input, demodulation folded into output scaling.
// ---------------------------------------------------------------------------

typedef __attribute__((ext_vector_type(16))) __bf16 v16bf;
typedef __attribute__((ext_vector_type(8)))  __bf16 v8bf;
typedef __attribute__((ext_vector_type(8)))  float  v8f;
typedef __attribute__((ext_vector_type(4)))  unsigned int u32x4;

#define NB    8
#define CIN   512
#define COUT  512
#define H_    64
#define W_    64
#define HW    (H_ * W_)          // 4096
#define KKTOT (CIN * 9)          // 4608 (GEMM K)
#define PAD   66                 // 64 + 1-px halo each side
#define PADSQ (PAD * PAD)        // 4356

// GEMM tiling
#define BM  128
#define BN  128                  // two image rows per N-tile
#define BK  32                   // bf16 WMMA K
#define NKT (KKTOT / BK)         // 144 = 9 k-taps x 16 slabs
#define AS  40                   // LDS row stride (halfs): 80B, 16B-aligned, bank-spread
#define BS  40                   // LDS col stride (halfs) for transposed B tile

// --------------------------------------------------------------------------
// 1) wbf[o][k*512+i] = bf16(gain * w[o][i][k]);  S_T[i][o] = gain^2*sum_k w^2
// --------------------------------------------------------------------------
__global__ __launch_bounds__(256) void prep_weight_k(const float* __restrict__ w,
                                                     __bf16* __restrict__ wbf,
                                                     float* __restrict__ S_T) {
    int t = blockIdx.x * 256 + threadIdx.x;   // 0 .. 512*512-1
    int o = t >> 9;
    int i = t & 511;
    const float gain = 1.0f / sqrtf((float)KKTOT);
    size_t src = (size_t)o * KKTOT + (size_t)i * 9;     // OIHW flat
    float s2 = 0.f;
#pragma unroll
    for (int k = 0; k < 9; ++k) {
        float v = w[src + k] * gain;
        s2 += v * v;
        wbf[(size_t)o * KKTOT + k * CIN + i] = (__bf16)v;   // k-major K layout
    }
    S_T[(size_t)i * COUT + o] = s2;
}

// --------------------------------------------------------------------------
// 2) sig[b][o] = rsqrt( sum_i S_T[i][o] * style[b,i]^2 + eps )
// --------------------------------------------------------------------------
__global__ __launch_bounds__(256) void calc_sigma_k(const float* __restrict__ S_T,
                                                    const float* __restrict__ style,
                                                    float* __restrict__ sig) {
    int t = blockIdx.x * 256 + threadIdx.x;   // 0 .. 4095
    int b = t >> 9;
    int o = t & 511;
    float acc = 0.f;
    for (int i = 0; i < CIN; ++i) {
        float s = style[b * CIN + i];
        acc += S_T[(size_t)i * COUT + o] * (s * s);
    }
    sig[t] = rsqrtf(acc + 1e-8f);
}

// --------------------------------------------------------------------------
// 3) xbp[b*CIN+i][66][66] = bf16(x * style), zero halo (branch-free im2col)
// --------------------------------------------------------------------------
__global__ __launch_bounds__(256) void modulate_pad_k(const float* __restrict__ x,
                                                      const float* __restrict__ style,
                                                      __bf16* __restrict__ xbp) {
    int bi = blockIdx.x;                 // b*CIN + i
    int b  = bi >> 9;
    int i  = bi & 511;
    float s = style[b * CIN + i];
    const float* src = x + (size_t)bi * HW;
    __bf16* dst = xbp + (size_t)bi * PADSQ;
    for (int e = threadIdx.x; e < PADSQ; e += 256) {
        int yy = e / PAD;
        int xx = e - yy * PAD;
        int yi = yy - 1, xi = xx - 1;
        float v = 0.f;
        if ((unsigned)yi < (unsigned)H_ && (unsigned)xi < (unsigned)W_)
            v = src[yi * W_ + xi] * s;
        dst[e] = (__bf16)v;
    }
}

// --------------------------------------------------------------------------
// 4) Implicit-GEMM conv with v_wmma_f32_16x16x32_bf16.
//    Grid (32 row-pairs, 4 M-tiles, 8 batches); block 256 = 8 waves (4x2),
//    wave tile 32x64 = 2x4 WMMA tiles; wn selects the image row.
//    Double-buffered LDS, 1 barrier / K-slab, global->reg prefetch of slab
//    kt+1 overlaps the 8 WMMAs of slab kt.
// --------------------------------------------------------------------------
__global__ __launch_bounds__(256) void modconv_gemm_k(const __bf16* __restrict__ wbf,
                                                      const __bf16* __restrict__ xbp,
                                                      const float*  __restrict__ sig,
                                                      float* __restrict__ out) {
    __shared__ __align__(16) __bf16 Als[2][BM * AS];   // A tiles: [128 rows][40]
    __shared__ __align__(16) __bf16 Bls[2][BN * BS];   // B tiles transposed: [128 cols][40]

    const int y0   = blockIdx.x * 2;      // first image row of the pair
    const int m0   = blockIdx.y * BM;     // Cout tile base
    const int b    = blockIdx.z;
    const int tid  = threadIdx.x;
    const int lane = tid & 31;
    const int wave = tid >> 5;            // 0..7
    const int wm   = wave >> 1;           // 0..3 (M: 32 rows each)
    const int wn   = wave & 1;            // 0..1 (N: image row y0+wn)

    const unsigned short* xsp = (const unsigned short*)xbp;

    // ---- per-thread constant address bases (hoisted out of hot loop) ----
    // A: 128x32 bf16 = 512 x 16B chunks; 2 chunks / thread
    const int arow0 = tid >> 2;                 // rows 0..63
    const int arow1 = (tid + 256) >> 2;         // rows 64..127
    const int aseg  = (tid & 3) * 8;            // 8-half segment
    const __bf16* pA0 = wbf + (size_t)(m0 + arow0) * KKTOT + aseg;
    const __bf16* pA1 = wbf + (size_t)(m0 + arow1) * KKTOT + aseg;

    // B: 32 i-rows x 128 px (2 image rows); 8 passes, lane = contiguous x-pair
    const int x2 = lane * 2;
    unsigned bBase[8];
#pragma unroll
    for (int p = 0; p < 8; ++p) {
        int rr    = (p & 3) * 8 + wave;         // i-row within slab: 0..31
        int chalf = p >> 2;                     // which image row of the pair
        bBase[p] = (unsigned)((b * CIN + rr) * PADSQ + chalf * PAD + x2);
    }

    // LDS fragment offsets (halfs), per-thread constants
    const int mr    = lane & 15;
    const int kb    = (lane >> 4) * 8;
    const int kb16  = (lane >> 4) * 16;
    const int aOff[2] = { (wm * 32 +      mr) * AS + kb,
                          (wm * 32 + 16 + mr) * AS + kb };
    int bOff[4];
#pragma unroll
    for (int tn = 0; tn < 4; ++tn)
        bOff[tn] = (wn * 64 + tn * 16 + mr) * BS + kb;

    u32x4    aPre[2];
    unsigned bPre[8];

    auto prefetch = [&](int kt) {
        const int kk0 = kt * BK;                    // offset in k-major K layout
        aPre[0] = *(const u32x4*)(pA0 + kk0);
        aPre[1] = *(const u32x4*)(pA1 + kk0);
        const int k  = kt >> 4;                     // uniform: conv tap 0..8
        const int ky = k / 3;
        const int kx = k - 3 * ky;
        const unsigned u = (unsigned)((kt & 15) * (32 * PADSQ) + (y0 + ky) * PAD + kx);
        if ((kx & 1) == 0) {                        // uniform branch: 4B-aligned pairs
#pragma unroll
            for (int p = 0; p < 8; ++p)
                bPre[p] = *(const unsigned*)(xsp + bBase[p] + u);
        } else {
#pragma unroll
            for (int p = 0; p < 8; ++p) {
                const unsigned short* s = xsp + bBase[p] + u;
                bPre[p] = (unsigned)s[0] | ((unsigned)s[1] << 16);
            }
        }
    };

    auto stage = [&](int buf) {
        *(u32x4*)&Als[buf][arow0 * AS + aseg] = aPre[0];
        *(u32x4*)&Als[buf][arow1 * AS + aseg] = aPre[1];
        unsigned short* bl = (unsigned short*)&Bls[buf][0];
#pragma unroll
        for (int p = 0; p < 8; ++p) {
            int rr  = (p & 3) * 8 + wave;
            int col = (p >> 2) * 64 + x2;
            bl[(col)     * BS + rr] = (unsigned short)(bPre[p] & 0xffffu);
            bl[(col + 1) * BS + rr] = (unsigned short)(bPre[p] >> 16);
        }
    };

    v8f acc[2][4] = {};
    union Frag { v16bf v; v8bf h[2]; };

    prefetch(0);
    stage(0);
    __syncthreads();

    for (int kt = 0; kt < NKT; ++kt) {
        const int cur = kt & 1;
        if (kt + 1 < NKT) prefetch(kt + 1);     // global loads overlap compute

        Frag a[2], bb[4];
#pragma unroll
        for (int tm = 0; tm < 2; ++tm) {
            a[tm].h[0] = *(const v8bf*)&Als[cur][aOff[tm]];        // K = kb..kb+7
            a[tm].h[1] = *(const v8bf*)&Als[cur][aOff[tm] + 16];   // K = 16+kb..
        }
#pragma unroll
        for (int tn = 0; tn < 4; ++tn) {
            bb[tn].h[0] = *(const v8bf*)&Bls[cur][bOff[tn]];       // K = kb..kb+7
            bb[tn].h[1] = *(const v8bf*)&Bls[cur][bOff[tn] + 16];
        }
#pragma unroll
        for (int tm = 0; tm < 2; ++tm)
#pragma unroll
            for (int tn = 0; tn < 4; ++tn)
                acc[tm][tn] = __builtin_amdgcn_wmma_f32_16x16x32_bf16(
                    false, a[tm].v, false, bb[tn].v,
                    (short)0, acc[tm][tn], false, false);

        if (kt + 1 < NKT) {
            stage(cur ^ 1);                     // waits loads, writes other buffer
            __syncthreads();
        }
    }

    // Epilogue: demodulate (row scale) + store fp32; wave owns image row y0+wn
    const int ncol = lane & 15;
    const int mhi  = (lane >> 4) * 8;
    const int yimg = y0 + wn;
#pragma unroll
    for (int tm = 0; tm < 2; ++tm) {
        v8f sv = *(const v8f*)(sig + b * COUT + m0 + wm * 32 + tm * 16 + mhi);
#pragma unroll
        for (int tn = 0; tn < 4; ++tn) {
            int nn = tn * 16 + ncol;                    // 0..63 -> x
#pragma unroll
            for (int v = 0; v < 8; ++v) {
                int o = m0 + wm * 32 + tm * 16 + mhi + v;
                out[(size_t)(b * COUT + o) * HW + yimg * W_ + nn] = acc[tm][tn][v] * sv[v];
            }
        }
    }
}

// --------------------------------------------------------------------------
extern "C" void kernel_launch(void* const* d_in, const int* in_sizes, int n_in,
                              void* d_out, int out_size, void* d_ws, size_t ws_size,
                              hipStream_t stream) {
    const float* fm    = (const float*)d_in[0];   // (8,512,64,64)
    const float* style = (const float*)d_in[1];   // (8,512)
    const float* w     = (const float*)d_in[2];   // (512,512,3,3)
    float* out = (float*)d_out;                   // (8,512,64,64)

    char* ws = (char*)d_ws;
    const size_t WBF_B = (size_t)COUT * KKTOT * 2;          // 4,718,592
    const size_t ST_B  = (size_t)CIN * COUT * 4;            // 1,048,576
    const size_t SIG_B = (size_t)NB * COUT * 4;             //    16,384
    __bf16* wbf = (__bf16*)(ws);
    float*  S_T = (float*)(ws + WBF_B);
    float*  sg  = (float*)(ws + WBF_B + ST_B);
    __bf16* xbp = (__bf16*)(ws + WBF_B + ST_B + SIG_B);     // 35,684,352 B

    prep_weight_k <<<(COUT * CIN) / 256, 256, 0, stream>>>(w, wbf, S_T);
    calc_sigma_k  <<<(NB * COUT) / 256,  256, 0, stream>>>(S_T, style, sg);
    modulate_pad_k<<<NB * CIN,           256, 0, stream>>>(fm, style, xbp);
    modconv_gemm_k<<<dim3(H_ / 2, COUT / BM, NB), 256, 0, stream>>>(wbf, xbp, sg, out);
}
